// CoAttentionLayer_8589934611
// MI455X (gfx1250) — matrix operand order repaired and checked by reading, not compiled, not run on hardware
//
#include <hip/hip_runtime.h>
#include <hip/hip_bf16.h>
#include <math.h>

typedef __attribute__((ext_vector_type(16))) __bf16 v16bf;
typedef __attribute__((ext_vector_type(8)))  __bf16 v8bf;
typedef __attribute__((ext_vector_type(4)))  __bf16 v4bf;
typedef __attribute__((ext_vector_type(8)))  float  v8f;
typedef __attribute__((ext_vector_type(4)))  int    v4i;

#define TM 64
#define TN 64
#define TK 64
#define KPAD 8
#define LDK (TK + KPAD)        // 72 elements -> 144B row stride, 16B aligned
#define GEMM_THREADS 128
#define SM_THREADS 256

// ---------------------------------------------------------------------------
// global -> LDS 16-byte copy: async-to-LDS on toolchains that expose the
// gfx1250 builtin, plain b128 load/store otherwise.
// Diagnostic from the probe round shows param 0 is `int4 __device__*` (AS1);
// param 1 is the LDS address (AS3, 32-bit pointer).
// ---------------------------------------------------------------------------
#if defined(__HIP_DEVICE_COMPILE__) && __has_builtin(__builtin_amdgcn_global_load_async_to_lds_b128)
#define HAVE_ASYNC_LDS 1
typedef __attribute__((address_space(1))) v4i as1_v4i;
typedef __attribute__((address_space(3))) v4i as3_v4i;
__device__ __forceinline__ void copy16_g2l(__bf16* dst, const __bf16* src) {
  __builtin_amdgcn_global_load_async_to_lds_b128(
      (as1_v4i*)(uintptr_t)src,
      (as3_v4i*)(unsigned int)(uintptr_t)dst, 0, 0);
}
__device__ __forceinline__ void async_copy_fence() {
#if __has_builtin(__builtin_amdgcn_s_wait_asynccnt)
  __builtin_amdgcn_s_wait_asynccnt(0);
#else
  asm volatile("s_wait_asynccnt 0" ::: "memory");
#endif
}
#else
#define HAVE_ASYNC_LDS 0
__device__ __forceinline__ void copy16_g2l(__bf16* dst, const __bf16* src) {
  *(v8bf*)dst = *(const v8bf*)src;
}
__device__ __forceinline__ void async_copy_fence() {}
#endif

// ---------------------------------------------------------------------------
// Batched transpose + f32->bf16 convert:  out[c][r] = (bf16) in[r][c]
// 32x32 tiles through LDS; block (32,8).
// ---------------------------------------------------------------------------
__global__ __launch_bounds__(256)
void transpose_cvt_kernel(const float* __restrict__ in, __bf16* __restrict__ out,
                          int R, int C, long long sIn, long long sOut) {
  __shared__ float tile[32][33];
  const long long bi = (long long)blockIdx.z * sIn;
  const long long bo = (long long)blockIdx.z * sOut;
  const int r0 = blockIdx.y * 32, c0 = blockIdx.x * 32;
  const int tx = threadIdx.x, ty = threadIdx.y;
#pragma unroll
  for (int i = 0; i < 32; i += 8)
    tile[ty + i][tx] = in[bi + (long long)(r0 + ty + i) * C + (c0 + tx)];
  __syncthreads();
#pragma unroll
  for (int i = 0; i < 32; i += 8)
    out[bo + (long long)(c0 + ty + i) * R + (r0 + tx)] = (__bf16)tile[tx][ty + i];
}

// ---------------------------------------------------------------------------
// Tiled NT GEMM via v_wmma_f32_16x16x32_bf16.
//   C[M,N] = A[M,K] * B^T      B is [N,K] row-major bf16 (always)
//   A: f32 (converted while staging) or bf16, row-major, lda = K
//   EPI_RELU_BF16: C = (bf16) relu(acc + bias[n]);  else C = f32 acc
// Block = 128 threads = 4 waves; 64x64 tile; wave owns a 32x32 quadrant as a
// 2x2 grid of 16x16 fragments; K staged 64 at a time (2 WMMA k-steps/barrier).
// ---------------------------------------------------------------------------
template <bool A_F32, bool EPI_RELU_BF16>
__global__ __launch_bounds__(GEMM_THREADS)
void gemm_wmma_kernel(const void* __restrict__ Ap, const __bf16* __restrict__ Bp,
                      const float* __restrict__ bias, void* __restrict__ Cp,
                      int M, int N, int K,
                      long long batchA, long long batchB, long long batchC) {
  __shared__ __bf16 As[TM][LDK];
  __shared__ __bf16 Bs[TN][LDK];

  const int tid = threadIdx.x;
  const int m0 = blockIdx.y * TM;
  const int n0 = blockIdx.x * TN;

  const float*  Af = (const float*)Ap;
  const __bf16* Ah = (const __bf16*)Ap;

  const long long aoff = (long long)blockIdx.z * batchA;
  const long long boff = (long long)blockIdx.z * batchB;
  const long long coff = (long long)blockIdx.z * batchC;

  const int lane  = tid & 31;
  const int wv    = tid >> 5;        // wave 0..3
  const int wm    = (wv >> 1) << 5;  // 0 or 32
  const int wn    = (wv & 1) << 5;   // 0 or 32
  const int lrow  = lane & 15;
  const int lhalf = lane >> 4;

  v8f acc[2][2] = {{{}, {}}, {{}, {}}};

  for (int k0 = 0; k0 < K; k0 += TK) {
    // --- prefetch next K tile of A (global_prefetch_b8) ---
    if (k0 + TK < K) {
      if (A_F32) __builtin_prefetch(&Af[aoff + (long long)(m0 + (tid >> 1)) * K + k0 + TK], 0, 1);
      else       __builtin_prefetch(&Ah[aoff + (long long)(m0 + (tid >> 1)) * K + k0 + TK], 0, 1);
    }

    // --- stage A tile [TM][TK] ---
    if (A_F32) {
      // 64 rows x 16 float4-chunks = 1024 chunks, 8 per thread
#pragma unroll
      for (int it = 0; it < (TM * TK / 4) / GEMM_THREADS; ++it) {
        int chunk = tid + it * GEMM_THREADS;
        int r  = chunk >> 4;
        int cc = (chunk & 15) * 4;
        const float4 v = *(const float4*)&Af[aoff + (long long)(m0 + r) * K + k0 + cc];
        v4bf h;
        h[0] = (__bf16)v.x; h[1] = (__bf16)v.y; h[2] = (__bf16)v.z; h[3] = (__bf16)v.w;
        *(v4bf*)&As[r][cc] = h;   // aligned ds_store_b64
      }
    } else {
      // 64 rows x 8 16B-chunks = 512 chunks, 4 per thread
#pragma unroll
      for (int it = 0; it < (TM * TK / 8) / GEMM_THREADS; ++it) {
        int chunk = tid + it * GEMM_THREADS;
        int r  = chunk >> 3;
        int cc = (chunk & 7) * 8;
        copy16_g2l(&As[r][cc], &Ah[aoff + (long long)(m0 + r) * K + k0 + cc]);
      }
    }
    // --- stage B tile [TN][TK] (B is [N,K] row-major bf16) ---
#pragma unroll
    for (int it = 0; it < (TN * TK / 8) / GEMM_THREADS; ++it) {
      int chunk = tid + it * GEMM_THREADS;
      int r  = chunk >> 3;
      int cc = (chunk & 7) * 8;
      copy16_g2l(&Bs[r][cc], &Bp[boff + (long long)(n0 + r) * K + k0 + cc]);
    }
    async_copy_fence();
    __syncthreads();

    // --- two 16x16x32 k-steps per stage ---
#pragma unroll
    for (int kk = 0; kk < TK; kk += 32) {
      v16bf afrag[2], bfrag[2];
#pragma unroll
      for (int mi = 0; mi < 2; ++mi) {
        const __bf16* arow = &As[wm + mi * 16 + lrow][kk];
#pragma unroll
        for (int e = 0; e < 16; ++e) {
          int p = e >> 1, q = e & 1;
          int k = (p < 4) ? (lhalf * 8 + 2 * p + q)
                          : (16 + lhalf * 8 + 2 * (p - 4) + q);
          afrag[mi][e] = arow[k];
        }
      }
#pragma unroll
      for (int ni = 0; ni < 2; ++ni) {
        const __bf16* brow = &Bs[wn + ni * 16 + lrow][kk];
#pragma unroll
        for (int e = 0; e < 16; ++e) bfrag[ni][e] = brow[(e < 8) ? (lhalf * 8 + e) : (16 + lhalf * 8 + (e - 8))]; // two 16B chunks -> ds_load_b128 x2
      }
#pragma unroll
      for (int mi = 0; mi < 2; ++mi)
#pragma unroll
        for (int ni = 0; ni < 2; ++ni)
          acc[mi][ni] = __builtin_amdgcn_wmma_f32_16x16x32_bf16(
              false, afrag[mi], false, bfrag[ni], (short)0, acc[mi][ni],
              false, false);
    }
    __syncthreads();
  }

  // --- epilogue (C/D layout: VGPR v -> row v | v+8 by lane half) ---
#pragma unroll
  for (int mi = 0; mi < 2; ++mi) {
#pragma unroll
    for (int ni = 0; ni < 2; ++ni) {
#pragma unroll
      for (int v = 0; v < 8; ++v) {
        int row = m0 + wm + mi * 16 + lhalf * 8 + v;
        int col = n0 + wn + ni * 16 + lrow;
        float x = acc[mi][ni][v];
        if constexpr (EPI_RELU_BF16) {
          x = fmaxf(x + bias[col], 0.0f);
          ((__bf16*)Cp)[coff + (long long)row * N + col] = (__bf16)x;
        } else {
          ((float*)Cp)[coff + (long long)row * N + col] = x;
        }
      }
    }
  }
}

// ---------------------------------------------------------------------------
// Masked row softmax: S[b,i,j] = softmax_j( E[b,i,j] - (1-mask[b,j])*NEG )
// ---------------------------------------------------------------------------
__global__ __launch_bounds__(SM_THREADS)
void row_softmax_mask_kernel(const float* __restrict__ E,
                             const int* __restrict__ mask,
                             __bf16* __restrict__ S, int L) {
  __shared__ float red[SM_THREADS];
  const int i = blockIdx.x;
  const int b = blockIdx.y;
  const long long base = ((long long)b * L + i) * (long long)L;
  const int* mrow = mask + (long long)b * L;
  const int t = threadIdx.x;
  const int PER = L / SM_THREADS;   // 4 for L=1024

  float vals[8];
  float mx = -3.0e38f;
  for (int p = 0; p < PER; ++p) {
    int j = t + p * SM_THREADS;
    float m = (float)mrow[j];
    float v = E[base + j] - (1.0f - m) * 1.0e30f;
    vals[p] = v;
    mx = fmaxf(mx, v);
  }
  red[t] = mx;
  __syncthreads();
  for (int s = SM_THREADS / 2; s > 0; s >>= 1) {
    if (t < s) red[t] = fmaxf(red[t], red[t + s]);
    __syncthreads();
  }
  mx = red[0];
  __syncthreads();

  float sum = 0.0f;
  for (int p = 0; p < PER; ++p) {
    vals[p] = __expf(vals[p] - mx);
    sum += vals[p];
  }
  red[t] = sum;
  __syncthreads();
  for (int s = SM_THREADS / 2; s > 0; s >>= 1) {
    if (t < s) red[t] += red[t + s];
    __syncthreads();
  }
  float inv = 1.0f / red[0];
  for (int p = 0; p < PER; ++p) {
    int j = t + p * SM_THREADS;
    S[base + j] = (__bf16)(vals[p] * inv);
  }
}

// ---------------------------------------------------------------------------
// host orchestration
// ---------------------------------------------------------------------------
extern "C" void kernel_launch(void* const* d_in, const int* in_sizes, int n_in,
                              void* d_out, int out_size, void* d_ws, size_t ws_size,
                              hipStream_t stream) {
  const float* x1   = (const float*)d_in[0];
  const float* x2   = (const float*)d_in[1];
  const float* kern = (const float*)d_in[2];   // [2, D, U]
  const float* bias = (const float*)d_in[3];   // [2, U]
  const int*  mask1 = (const int*)d_in[4];
  const int*  mask2 = (const int*)d_in[5];

  const int Bn = 16, L = 1024, D = 512, U = 512;
  const long long BLD = (long long)Bn * L * D;
  const long long BLL = (long long)Bn * L * L;
  const long long DU  = (long long)D * U;

  // workspace layout (~161 MiB): everything a GEMM reads is bf16 and NT-ready
  __bf16* x1T = (__bf16*)d_ws;        // [B, D, L]
  __bf16* x2T = x1T + BLD;            // [B, D, L]
  __bf16* W0T = x2T + BLD;            // [U, D]
  __bf16* W1T = W0T + DU;             // [U, D]
  __bf16* Q1b = W1T + DU;             // [B, L, U]  (reused for P1)
  __bf16* Q2b = Q1b + BLD;            // [B, L, U]  (reused for P2)
  float*  E   = (float*)(Q2b + BLD);  // [B, L, L] f32 scores (reused 4x)
  __bf16* S   = (__bf16*)(E + BLL);   // [B, L, L] bf16 softmax weights

  float* beta  = (float*)d_out;
  float* alpha = beta + BLD;
  float* q1n   = alpha + BLD;
  float* q2n   = q1n + BLD;

  dim3 blk(GEMM_THREADS);
  dim3 tblk(32, 8);
  const long long sQ = (long long)L * U;
  const long long sE = (long long)L * L;
  const long long sX = (long long)D * L;   // per-batch stride of x^T

  // 0) pre-transpose operands to bf16 NT form
  transpose_cvt_kernel<<<dim3(D / 32, L / 32, Bn), tblk, 0, stream>>>(
      x1, x1T, L, D, (long long)L * D, sX);
  transpose_cvt_kernel<<<dim3(D / 32, L / 32, Bn), tblk, 0, stream>>>(
      x2, x2T, L, D, (long long)L * D, sX);
  transpose_cvt_kernel<<<dim3(U / 32, D / 32, 1), tblk, 0, stream>>>(
      kern, W0T, D, U, 0, 0);
  transpose_cvt_kernel<<<dim3(U / 32, D / 32, 1), tblk, 0, stream>>>(
      kern + DU, W1T, D, U, 0, 0);

  // 1) Q1 = relu(x1*W0 + b0), Q2 = relu(x2*W0 + b0)
  {
    dim3 g(U / TN, (Bn * L) / TM, 1);
    gemm_wmma_kernel<true, true><<<g, blk, 0, stream>>>(
        x1, W0T, bias, Q1b, Bn * L, U, D, 0, 0, 0);
    gemm_wmma_kernel<true, true><<<g, blk, 0, stream>>>(
        x2, W0T, bias, Q2b, Bn * L, U, D, 0, 0, 0);
  }

  // 2) E = Q1*Q2^T ; beta = softmax_j(E, mask2) * x2
  {
    dim3 ge(L / TN, L / TM, Bn);
    gemm_wmma_kernel<false, false><<<ge, blk, 0, stream>>>(
        Q1b, Q2b, nullptr, E, L, L, U, sQ, sQ, sE);
    row_softmax_mask_kernel<<<dim3(L, Bn), SM_THREADS, 0, stream>>>(E, mask2, S, L);
    dim3 g(D / TN, L / TM, Bn);
    gemm_wmma_kernel<false, false><<<g, blk, 0, stream>>>(
        S, x2T, nullptr, beta, L, D, L, sE, sX, (long long)L * D);
  }

  // 3) E^T = Q2*Q1^T ; alpha = softmax_i(E^T, mask1) * x1
  {
    dim3 ge(L / TN, L / TM, Bn);
    gemm_wmma_kernel<false, false><<<ge, blk, 0, stream>>>(
        Q2b, Q1b, nullptr, E, L, L, U, sQ, sQ, sE);
    row_softmax_mask_kernel<<<dim3(L, Bn), SM_THREADS, 0, stream>>>(E, mask1, S, L);
    dim3 g(D / TN, L / TM, Bn);
    gemm_wmma_kernel<false, false><<<g, blk, 0, stream>>>(
        S, x1T, nullptr, alpha, L, D, L, sE, sX, (long long)L * D);
  }

  // 4) P1 = relu(x1*W1 + b1) -> Q1b ; P2 -> Q2b
  {
    dim3 g(U / TN, (Bn * L) / TM, 1);
    gemm_wmma_kernel<true, true><<<g, blk, 0, stream>>>(
        x1, W1T, bias + U, Q1b, Bn * L, U, D, 0, 0, 0);
    gemm_wmma_kernel<true, true><<<g, blk, 0, stream>>>(
        x2, W1T, bias + U, Q2b, Bn * L, U, D, 0, 0, 0);
  }

  // 5) F1 = P1*P1^T ; Q1_new = softmax_j(F1, mask1) * x1
  {
    dim3 ge(L / TN, L / TM, Bn);
    gemm_wmma_kernel<false, false><<<ge, blk, 0, stream>>>(
        Q1b, Q1b, nullptr, E, L, L, U, sQ, sQ, sE);
    row_softmax_mask_kernel<<<dim3(L, Bn), SM_THREADS, 0, stream>>>(E, mask1, S, L);
    dim3 g(D / TN, L / TM, Bn);
    gemm_wmma_kernel<false, false><<<g, blk, 0, stream>>>(
        S, x1T, nullptr, q1n, L, D, L, sE, sX, (long long)L * D);
  }

  // 6) F2 = P2*P2^T ; Q2_new = softmax_j(F2, mask2) * x2
  {
    dim3 ge(L / TN, L / TM, Bn);
    gemm_wmma_kernel<false, false><<<ge, blk, 0, stream>>>(
        Q2b, Q2b, nullptr, E, L, L, U, sQ, sQ, sE);
    row_softmax_mask_kernel<<<dim3(L, Bn), SM_THREADS, 0, stream>>>(E, mask2, S, L);
    dim3 g(D / TN, L / TM, Bn);
    gemm_wmma_kernel<false, false><<<g, blk, 0, stream>>>(
        S, x2T, nullptr, q2n, L, D, L, sE, sX, (long long)L * D);
  }
}
